// BezierDecoder_3977139716916
// MI455X (gfx1250) — matrix operand order repaired and checked by reading, not compiled, not run on hardware
//
#include <hip/hip_runtime.h>
#include <hip/hip_bf16.h>
#include <math.h>

#define LATENT_DIM 128
#define HIDDEN 1024
#define NUM_PATHS 4
#define NUM_SEGMENTS 3
#define CANVAS 64
#define POINTS_PER_PATH 10
#define T_SAMPLES 16                         // SAMPLES_PER_SEG = 4*4
#define N_SAMPLES (NUM_SEGMENTS * T_SAMPLES) // 48 samples per closed path
#define BATCH 32
#define NPARAM 88                            // 80 points + 4 widths + 4 alphas

typedef __attribute__((ext_vector_type(16))) _Float16     v16h;
typedef __attribute__((ext_vector_type(8)))  float        v8f;
typedef __attribute__((ext_vector_type(4)))  unsigned int u32x4;
typedef __attribute__((ext_vector_type(8)))  int          i32x8;
typedef __attribute__((ext_vector_type(4)))  int          i32x4;

__device__ __forceinline__ float selu_f(float x) {
    const float scale = 1.0507009873554805f;
    const float alpha = 1.6732632423543772f;
    return scale * (x > 0.f ? x : alpha * (__expf(x) - 1.f));
}

// Low 32 bits of a generic pointer into LDS == LDS byte offset (shared aperture).
__device__ __forceinline__ unsigned lds_addr_of(const void* p) {
    return (unsigned)(uintptr_t)p;
}

// ---------------------------------------------------------------------------
// TDM: async-load a 32(row) x 16(col) tile of f32 from a row-major [*, N]
// matrix into LDS, packed contiguously (row stride 16 floats).
// D# per CDNA5 ISA:
//   group0: count=1 | lds_addr | global_addr[56:0] | type=2
//   group1: data_size=2(4B) | tensor_dim0=N | tensor_dim1=32
//           | tile_dim0=16 | tile_dim1=32 | tensor_dim0_stride=N
//   groups 2/3: zero (2D tensor)
// Tracked with TENSORcnt.
// ---------------------------------------------------------------------------
__device__ __forceinline__ void tdm_load_tile_b(const void* gptr, unsigned lds_off,
                                                unsigned n_stride) {
    const unsigned long long ga = (unsigned long long)(uintptr_t)gptr;
    u32x4 g0 = { 1u,                                     // count=1 (valid user D#)
                 lds_off,                                // lds_addr
                 (unsigned)ga,                           // global_addr[31:0]
                 ((unsigned)(ga >> 32) & 0x01FFFFFFu)    // global_addr[56:32]
                     | 0x80000000u };                    // type=2 ("image")
    i32x8 g1;
    g1[0] = 0x00020000;                                  // data_size=4B, mask=0
    g1[1] = (int)((n_stride & 0xFFFFu) << 16);           // tensor_dim0 lo16 @bit48
    g1[2] = (int)((n_stride >> 16) | (32u << 16));       // tensor_dim0 hi | tensor_dim1=32
    g1[3] = (int)(16u << 16);                            // tile_dim0=16 @bit112
    g1[4] = 32;                                          // tile_dim1=32 @bit128
    g1[5] = (int)n_stride;                               // tensor_dim0_stride lo32
    g1[6] = 0;                                           // stride hi | dim1_stride lo
    g1[7] = 0;
    i32x4 z4 = {0, 0, 0, 0};
#if defined(__clang_major__) && (__clang_major__ >= 23)
    i32x8 z8 = {0, 0, 0, 0, 0, 0, 0, 0};
    __builtin_amdgcn_tensor_load_to_lds(g0, g1, z4, z4, z8, 0);
#else
    __builtin_amdgcn_tensor_load_to_lds(g0, g1, z4, z4, 0);
#endif
}

// ---------------------------------------------------------------------------
// GEMM + bias + SELU via V_WMMA_F32_16X16X32_F16. One wave per 16x16 C tile.
// B tiles are DMA'd into LDS by the Tensor Data Mover and double-buffered on
// TENSORcnt; A fragments load directly (tiny, L2-resident).
// Fragment layouts as built here (wave32; K contiguous per lane half, not ISA table order):
//   A 16x32 f16 : lanes 0-15 -> M=lane (K 0..15),
//                 lanes 16-31 -> M=lane-16 (K 16..31)
//   B 32x16 f16 : lanes 0-15 hold K=0..15, lanes 16-31 hold K=16..31; 2 K/VGPR
//   C/D f32     : M = vgpr + 8*(lane>=16), N = lane&15
// ---------------------------------------------------------------------------
template <int K>
__global__ void wmma_gemm_selu(const float* __restrict__ A,
                               const float* __restrict__ B,
                               const float* __restrict__ bias,
                               float* __restrict__ C,
                               int M, int N) {
    __shared__ float bt[2][4][32 * 16];            // 2 buffers x 4 waves x 2KB
    const int lane   = threadIdx.x & 31;
    const int wave   = threadIdx.x >> 5;
    const int tilesN = N >> 4;
    const int tilesM = (M + 15) >> 4;
    const int tile   = blockIdx.x * (blockDim.x >> 5) + wave;
    if (tile >= tilesM * tilesN) return;           // wave-uniform: EXEC stays full
    const int tm = tile / tilesN;
    const int tn = tile - tm * tilesN;
    const int m0 = tm << 4, n0 = tn << 4;
    const int half = lane >> 4;
    const int r    = lane & 15;

    const unsigned lds0 = lds_addr_of(&bt[0][wave][0]);
    const unsigned lds1 = lds_addr_of(&bt[1][wave][0]);

    // Prime the pipeline: first B tile in flight.
    tdm_load_tile_b(B + n0, lds0, (unsigned)N);

    v8f acc = {};
#pragma unroll 2
    for (int k0 = 0; k0 < K; k0 += 32) {
        const int buf = (k0 >> 5) & 1;
        if (k0 + 32 < K) {                         // issue next tile, keep 1 in flight
            tdm_load_tile_b(B + (size_t)(k0 + 32) * N + n0,
                            buf ? lds0 : lds1, (unsigned)N);
            __builtin_amdgcn_s_wait_tensorcnt((short)1);
        } else {
            __builtin_amdgcn_s_wait_tensorcnt((short)0);
        }
        asm volatile("" ::: "memory");             // keep ds_loads after the wait

        v16h a, b;
#pragma unroll
        for (int v = 0; v < 8; ++v) {
            // A fragment pair (ka, ka+1) for this lane/vgpr (global, f32->f16)
            const int ka = half * 16 + 2 * v;
            const float* ap = A + (size_t)(m0 + r) * K + k0 + ka;
            a[2 * v]     = (_Float16)ap[0];
            a[2 * v + 1] = (_Float16)ap[1];
            // B fragment from the LDS tile (row-packed 32x16)
            const int kb = half * 16 + 2 * v;
            b[2 * v]     = (_Float16)bt[buf][wave][kb * 16 + r];
            b[2 * v + 1] = (_Float16)bt[buf][wave][(kb + 1) * 16 + r];
        }
        acc = __builtin_amdgcn_wmma_f32_16x16x32_f16(
            /*neg_a=*/false, a, /*neg_b=*/false, b,
            /*c_mod=*/(short)0, acc, /*reuse_a=*/false, /*reuse_b=*/false);
    }
#pragma unroll
    for (int v = 0; v < 8; ++v) {
        const int mo = m0 + v + 8 * half;
        const int no = n0 + r;
        if (mo < M) C[(size_t)mo * N + no] = selu_f(acc[v] + bias[no]);
    }
}

// ---------------------------------------------------------------------------
// Heads: points = tanh(h@Wp+bp)*30+32 ; widths = sigmoid(h@Ww+bw)*2+1 ;
// alphas = sigmoid(h@Wa+ba). params[b][0..79|80..83|84..87].
// ---------------------------------------------------------------------------
__global__ void heads_kernel(const float* __restrict__ h,
                             const float* __restrict__ Wp, const float* __restrict__ bp,
                             const float* __restrict__ Ww, const float* __restrict__ bw,
                             const float* __restrict__ Wa, const float* __restrict__ ba,
                             float* __restrict__ params) {
    const int bI = blockIdx.x;
    const int j  = threadIdx.x;
    if (j >= NPARAM) return;
    const float* hrow = h + (size_t)bI * HIDDEN;
    const float* w; float bias0; int ldw;
    if (j < 80)      { w = Wp + j;        bias0 = bp[j];      ldw = 80; }
    else if (j < 84) { w = Ww + (j - 80); bias0 = bw[j - 80]; ldw = 4;  }
    else             { w = Wa + (j - 84); bias0 = ba[j - 84]; ldw = 4;  }
    float s = bias0;
    for (int k = 0; k < HIDDEN; ++k) s += hrow[k] * w[(size_t)k * ldw];
    float outv;
    if (j < 80)      outv = tanhf(s) * 30.0f + 32.0f;     // (CANVAS/2-2)*x + 32
    else if (j < 84) outv = 2.0f / (1.0f + __expf(-s)) + 1.0f;
    else             outv = 1.0f / (1.0f + __expf(-s));
    params[(size_t)bI * NPARAM + j] = outv;
}

// ---------------------------------------------------------------------------
// Sample cubic Bezier segments: pts[b][p][n] for n in [0,48), t = i/15.
// ---------------------------------------------------------------------------
__global__ void sample_kernel(const float* __restrict__ params,
                              float* __restrict__ pts) {
    const int idx = blockIdx.x * blockDim.x + threadIdx.x;
    if (idx >= BATCH * NUM_PATHS * N_SAMPLES) return;
    const int n  = idx % N_SAMPLES;
    const int p  = (idx / N_SAMPLES) % NUM_PATHS;
    const int bI = idx / (N_SAMPLES * NUM_PATHS);
    const int s  = n / T_SAMPLES;
    const int ti = n % T_SAMPLES;
    const float t = (float)ti * (1.0f / (float)(T_SAMPLES - 1));
    const float u = 1.0f - t;
    const float w0 = u * u * u, w1 = 3.f * t * u * u,
                w2 = 3.f * t * t * u, w3 = t * t * t;
    const float* pp = params + (size_t)bI * NPARAM
                    + p * POINTS_PER_PATH * 2 + s * 3 * 2;
    const float x = w0 * pp[0] + w1 * pp[2] + w2 * pp[4] + w3 * pp[6];
    const float y = w0 * pp[1] + w1 * pp[3] + w2 * pp[5] + w3 * pp[7];
    float* o = pts + ((size_t)(bI * NUM_PATHS + p) * N_SAMPLES + n) * 2;
    o[0] = x; o[1] = y;
}

// ---------------------------------------------------------------------------
// Renderer: per pixel, per path: min distance to polyline + winding number
// (sum of atan2 of consecutive diff cross/dot). Polylines staged in LDS.
// 256 pixels/block, 16 blocks per batch image.
// ---------------------------------------------------------------------------
__global__ void render_kernel(const float* __restrict__ params,
                              const float* __restrict__ pts,
                              float* __restrict__ out) {
    __shared__ float2 spts[NUM_PATHS][N_SAMPLES];
    __shared__ float  sw[NUM_PATHS], sa[NUM_PATHS];
    const int bI    = blockIdx.x >> 4;
    const int chunk = blockIdx.x & 15;
    const int tid   = threadIdx.x;
    const float* pb = pts + (size_t)bI * NUM_PATHS * N_SAMPLES * 2;
    for (int i = tid; i < NUM_PATHS * N_SAMPLES; i += blockDim.x)
        spts[i / N_SAMPLES][i % N_SAMPLES] = make_float2(pb[2 * i], pb[2 * i + 1]);
    if (tid < NUM_PATHS) {
        sw[tid] = params[(size_t)bI * NPARAM + 80 + tid];
        sa[tid] = params[(size_t)bI * NPARAM + 84 + tid];
    }
    __syncthreads();

    const int   pix = chunk * 256 + tid;
    const float px  = (float)(pix & 63) + 0.5f;
    const float py  = (float)(pix >> 6) + 0.5f;
    const float inv2pi = 0.15915494309189535f;

    float accp = 1.0f;
#pragma unroll
    for (int p = 0; p < NUM_PATHS; ++p) {
        float d0x = px - spts[p][0].x;
        float d0y = py - spts[p][0].y;
        const float fdx = d0x, fdy = d0y;         // wrap-around sample
        float dmin2 = d0x * d0x + d0y * d0y;
        float wind  = 0.0f;
        for (int n = 0; n < N_SAMPLES; ++n) {
            float d1x, d1y;
            if (n == N_SAMPLES - 1) { d1x = fdx; d1y = fdy; }
            else { float2 q = spts[p][n + 1]; d1x = px - q.x; d1y = py - q.y; }
            const float cr = d0x * d1y - d0y * d1x;
            const float dt = d0x * d1x + d0y * d1y;
            wind += atan2f(cr, dt);
            dmin2 = fminf(dmin2, d1x * d1x + d1y * d1y);
            d0x = d1x; d0y = d1y;
        }
        const float dmin   = sqrtf(dmin2 + 1e-8f);
        const float stroke = fminf(fmaxf(sw[p] * 0.5f + 0.5f - dmin, 0.f), 1.f);
        const float fill   = fminf(fabsf(wind * inv2pi), 1.f);
        const float cov    = fmaxf(stroke, fill);
        accp *= (1.0f - sa[p] * cov);
    }
    out[(size_t)bI * (CANVAS * CANVAS) + pix] = 1.0f - accp;
}

// ---------------------------------------------------------------------------
extern "C" void kernel_launch(void* const* d_in, const int* in_sizes, int n_in,
                              void* d_out, int out_size, void* d_ws, size_t ws_size,
                              hipStream_t stream) {
    (void)in_sizes; (void)n_in; (void)out_size; (void)ws_size;
    const float* z  = (const float*)d_in[0];
    const float* W1 = (const float*)d_in[1];
    const float* b1 = (const float*)d_in[2];
    const float* W2 = (const float*)d_in[3];
    const float* b2 = (const float*)d_in[4];
    const float* Wp = (const float*)d_in[5];
    const float* bp = (const float*)d_in[6];
    const float* Ww = (const float*)d_in[7];
    const float* bw = (const float*)d_in[8];
    const float* Wa = (const float*)d_in[9];
    const float* ba = (const float*)d_in[10];

    float* ws     = (float*)d_ws;
    float* h1     = ws;                              // 32*1024
    float* h2     = h1 + BATCH * HIDDEN;             // 32*1024
    float* params = h2 + BATCH * HIDDEN;             // 32*88
    float* ptsbuf = params + BATCH * NPARAM;         // 32*4*48*2

    // 128 tiles of 16x16, 4 waves/block -> 32 blocks per layer.
    {
        int tiles  = ((BATCH + 15) / 16) * (HIDDEN / 16);
        int blocks = (tiles + 3) / 4;
        wmma_gemm_selu<LATENT_DIM><<<blocks, 128, 0, stream>>>(z, W1, b1, h1,
                                                               BATCH, HIDDEN);
        wmma_gemm_selu<HIDDEN><<<blocks, 128, 0, stream>>>(h1, W2, b2, h2,
                                                           BATCH, HIDDEN);
    }
    heads_kernel<<<BATCH, 96, 0, stream>>>(h2, Wp, bp, Ww, bw, Wa, ba, params);
    sample_kernel<<<(BATCH * NUM_PATHS * N_SAMPLES + 255) / 256, 256, 0, stream>>>(
        params, ptsbuf);
    render_kernel<<<BATCH * 16, 256, 0, stream>>>(params, ptsbuf, (float*)d_out);
}
